// MultiHeadLocalAttention_687194768282
// MI455X (gfx1250) — matrix operand order repaired and checked by reading, not compiled, not run on hardware
//
#include <hip/hip_runtime.h>
#include <hip/hip_bf16.h>
#include <stdint.h>

#define S_LEN   4096
#define D_MODEL 1024
#define N_HEADS 16
#define HEAD_DIM 64
#define WIN     256

typedef __bf16 v16bf __attribute__((ext_vector_type(16)));
typedef float  v8f   __attribute__((ext_vector_type(8)));

union FragU { uint4 u[2]; v16bf v; };

__device__ __forceinline__ v8f zero8() {
  v8f z;
#pragma unroll
  for (int i = 0; i < 8; ++i) z[i] = 0.0f;
  return z;
}

// round-to-nearest-even f32 -> bf16 (bit pattern in ushort)
__device__ __forceinline__ unsigned short f2bf(float x) {
  unsigned u = __float_as_uint(x);
  u = (u + 0x7FFFu + ((u >> 16) & 1u)) >> 16;
  return (unsigned short)u;
}

// generic pointer to LDS byte offset (low 32 bits of flat LDS address)
__device__ __forceinline__ unsigned lds_off(const void* p) {
  return (unsigned)(uintptr_t)p;
}

// CDNA5 async DMA: global -> LDS, 16 bytes, tracked by ASYNCcnt
__device__ __forceinline__ void async_load_b128(unsigned lds_addr, const void* g) {
  unsigned long long ga = (unsigned long long)(uintptr_t)g;
  asm volatile("global_load_async_to_lds_b128 %0, %1, off"
               :: "v"(lds_addr), "v"(ga) : "memory");
}
__device__ __forceinline__ void wait_asynccnt0() {
  asm volatile("s_wait_asynccnt 0" ::: "memory");
}

// B-fragment (contiguous K halves): 32 contiguous bytes per lane
__device__ __forceinline__ v16bf ld_contig(const unsigned short* p) {
  FragU f;
  const uint4* q = (const uint4*)p;
  f.u[0] = q[0]; f.u[1] = q[1];
  return f.v;
}
// A-fragment: two 16B chunks (callers pass adjacent: K base..base+7 and K base+8..base+15)
__device__ __forceinline__ v16bf ld_split(const unsigned short* p0, const unsigned short* p1) {
  FragU f;
  f.u[0] = *(const uint4*)p0; f.u[1] = *(const uint4*)p1;
  return f.v;
}

// ---------------------------------------------------------------------------
// Kernel 1: pack x and [Wq;Wk;Wv] into bf16 (row-major, k contiguous)
// ---------------------------------------------------------------------------
__global__ __launch_bounds__(256) void pack_kernel(
    const float* __restrict__ x, const float* __restrict__ wq,
    const float* __restrict__ wk, const float* __restrict__ wv,
    unsigned short* __restrict__ xb, unsigned short* __restrict__ wb) {
  const long long idx = (long long)blockIdx.x * 256 + threadIdx.x;
  const long long NX = (long long)S_LEN * D_MODEL;
  const long long NW = (long long)D_MODEL * D_MODEL;
  if (idx < NX) { xb[idx] = f2bf(x[idx]); return; }
  long long j = idx - NX;
  if (j >= 3 * NW) return;
  const float* w = (j < NW) ? wq : ((j < 2 * NW) ? wk : wv);
  long long jl = (j < NW) ? j : ((j < 2 * NW) ? j - NW : j - 2 * NW);
  wb[j] = f2bf(w[jl]);
}

// ---------------------------------------------------------------------------
// Kernel 2: QKV projection GEMM.  out[s,j] = sum_k x[s,k]*W[j,k] + b[j]
// Block tile 128 rows x 64 cols, 8 waves, each wave a 2x2 WMMA register tile
// (32B of LDS fragment traffic per WMMA).  Async double-buffered LDS staging.
// Q scaled by 1/sqrt(64); V stored transposed [H][64][S] for the attention pass.
// ---------------------------------------------------------------------------
__global__ __launch_bounds__(256) void qkv_gemm_kernel(
    const unsigned short* __restrict__ xb, const unsigned short* __restrict__ wb,
    const float* __restrict__ bq, const float* __restrict__ bk,
    const float* __restrict__ bv,
    unsigned short* __restrict__ Qb, unsigned short* __restrict__ Kb,
    unsigned short* __restrict__ Vt) {
  __shared__ unsigned short lx[2][128 * 32];  // 8 KB per buffer
  __shared__ unsigned short lw[2][64 * 32];   // 4 KB per buffer
  const int tid = threadIdx.x, lane = tid & 31, wave = tid >> 5;
  const int ib = blockIdx.y * 128;        // row block (s)
  const int jb = blockIdx.x * 64;         // col block (0..3071 across Q,K,V)
  const int rt0 = (wave & 3) * 2;         // 2 row tiles per wave
  const int ct0 = (wave >> 2) * 2;        // 2 col tiles per wave
  const int m = lane & 15, hi = lane >> 4;
  const int base = hi * 8;                // unused (A-frag uses contiguous K halves via kb2)
  const int kb2 = hi * 16;                // A/B-frag K base (contiguous K halves, k = 16*hi + s)

  auto stage = [&](int ks, int buf) {
    const int kk = ks * 32;
    {  // x tile: 128 rows x 4 16B-chunks = 512 chunks, 2 per thread
      const int id0 = tid, id1 = tid + 256;
      async_load_b128(lds_off(&lx[buf][(id0 >> 2) * 32 + (id0 & 3) * 8]),
                      xb + (size_t)(ib + (id0 >> 2)) * D_MODEL + kk + (id0 & 3) * 8);
      async_load_b128(lds_off(&lx[buf][(id1 >> 2) * 32 + (id1 & 3) * 8]),
                      xb + (size_t)(ib + (id1 >> 2)) * D_MODEL + kk + (id1 & 3) * 8);
    }
    {  // W tile: 64 rows x 4 chunks = 256 chunks, 1 per thread
      const int r = tid >> 2, c = tid & 3;
      async_load_b128(lds_off(&lw[buf][r * 32 + c * 8]),
                      wb + (size_t)(jb + r) * D_MODEL + kk + c * 8);
    }
  };

  v8f acc00 = zero8(), acc01 = zero8(), acc10 = zero8(), acc11 = zero8();
  stage(0, 0);
  wait_asynccnt0();
  __syncthreads();

  for (int ks = 0; ks < 32; ++ks) {
    const int buf = ks & 1;
    if (ks + 1 < 32) stage(ks + 1, buf ^ 1);

    const unsigned short* xs = &lx[buf][0];
    const unsigned short* ws = &lw[buf][0];
    const unsigned short* ar0 = xs + ((rt0 + 0) * 16 + m) * 32;
    const unsigned short* ar1 = xs + ((rt0 + 1) * 16 + m) * 32;
    const v16bf a0 = ld_split(ar0 + kb2, ar0 + kb2 + 8);
    const v16bf a1 = ld_split(ar1 + kb2, ar1 + kb2 + 8);
    const v16bf b0 = ld_contig(ws + ((ct0 + 0) * 16 + m) * 32 + kb2);
    const v16bf b1 = ld_contig(ws + ((ct0 + 1) * 16 + m) * 32 + kb2);
    acc00 = __builtin_amdgcn_wmma_f32_16x16x32_bf16(false, a0, false, b0, (short)0, acc00, false, false);
    acc01 = __builtin_amdgcn_wmma_f32_16x16x32_bf16(false, a0, false, b1, (short)0, acc01, false, false);
    acc10 = __builtin_amdgcn_wmma_f32_16x16x32_bf16(false, a1, false, b0, (short)0, acc10, false, false);
    acc11 = __builtin_amdgcn_wmma_f32_16x16x32_bf16(false, a1, false, b1, (short)0, acc11, false, false);

    wait_asynccnt0();
    __syncthreads();
  }

  const int region = jb >> 10;            // 0=Q, 1=K, 2=V (block never straddles)
  const float* bias = (region == 0) ? bq : ((region == 1) ? bk : bv);
#pragma unroll
  for (int ti = 0; ti < 2; ++ti) {
#pragma unroll
    for (int tj = 0; tj < 2; ++tj) {
      const v8f acc = (ti == 0) ? (tj == 0 ? acc00 : acc01)
                                : (tj == 0 ? acc10 : acc11);
      const int jcol = jb + (ct0 + tj) * 16 + m;
      const int jl = jcol & (D_MODEL - 1);
      const int h = jl >> 6, d = jl & 63;
      const float bs = bias[jl];
#pragma unroll
      for (int r = 0; r < 8; ++r) {
        const int i = ib + (rt0 + ti) * 16 + r + hi * 8;
        const float val = acc[r] + bs;
        if (region == 0)
          Qb[((size_t)(h * S_LEN) + i) * HEAD_DIM + d] = f2bf(val * 0.125f);
        else if (region == 1)
          Kb[((size_t)(h * S_LEN) + i) * HEAD_DIM + d] = f2bf(val);
        else
          Vt[((size_t)(h * HEAD_DIM) + d) * S_LEN + i] = f2bf(val);
      }
    }
  }
}

// ---------------------------------------------------------------------------
// Kernel 3: banded flash attention.  Block = 8 waves = 128 queries of one head.
// Streams 32-key blocks through async double-buffered LDS (K rows + V^T rows).
// ---------------------------------------------------------------------------
__global__ __launch_bounds__(256) void attn_kernel(
    const unsigned short* __restrict__ Qb, const unsigned short* __restrict__ Kb,
    const unsigned short* __restrict__ Vt, float* __restrict__ out) {
  __shared__ unsigned short lk[2][32 * 64];  // 32 keys x 64 depth
  __shared__ unsigned short lv[2][64 * 32];  // 64 depth x 32 keys (V^T)
  __shared__ unsigned short lp[8][16 * 32];  // per-wave P bounce buffer
  const int tid = threadIdx.x, lane = tid & 31, wave = tid >> 5;
  const int h = blockIdx.y;
  const int qb0 = blockIdx.x * 128;
  const int iw = qb0 + wave * 16;
  const int m = lane & 15, hi = lane >> 4;
  const int base = hi * 8, kb2 = hi * 16;

  // Q A-fragments (already scaled by 1/sqrt(depth))
  const unsigned short* qrow = Qb + ((size_t)(h * S_LEN) + iw + m) * HEAD_DIM;
  const v16bf qa0 = ld_split(qrow + kb2, qrow + kb2 + 8);
  const v16bf qa1 = ld_split(qrow + 32 + kb2, qrow + 32 + kb2 + 8);

  const int j_lo = max(qb0 - WIN, 0);
  const int j_hi = min(qb0 + 127 + WIN, S_LEN - 1);
  const int kb_start = j_lo & ~31;
  const int nblocks = ((j_hi - kb_start) >> 5) + 1;

  auto stage = [&](int kb, int buf) {
    const int j0 = kb_start + kb * 32;
    { const int r = tid >> 3, c = tid & 7;   // 32 rows x 8 chunks
      async_load_b128(lds_off(&lk[buf][r * 64 + c * 8]),
                      Kb + ((size_t)(h * S_LEN) + j0 + r) * HEAD_DIM + c * 8); }
    { const int r = tid >> 2, c = tid & 3;   // 64 rows x 4 chunks
      async_load_b128(lds_off(&lv[buf][r * 32 + c * 8]),
                      Vt + ((size_t)(h * HEAD_DIM) + r) * S_LEN + j0 + c * 8); }
  };

  float rm[8], rs[8];
  v8f o0 = zero8(), o1 = zero8(), o2 = zero8(), o3 = zero8();
#pragma unroll
  for (int r = 0; r < 8; ++r) { rm[r] = -3.0e38f; rs[r] = 0.0f; }

  stage(0, 0);
  wait_asynccnt0();
  __syncthreads();

  for (int kb = 0; kb < nblocks; ++kb) {
    const int buf = kb & 1;
    const int j0 = kb_start + kb * 32;
    if (kb + 1 < nblocks) stage(kb + 1, buf ^ 1);

    // wave-uniform band check: keep EXEC all-ones around WMMA
    if (j0 + 31 >= iw - WIN && j0 <= iw + 15 + WIN) {
      const unsigned short* lkb = &lk[buf][0];
      const v16bf kf00 = ld_contig(lkb + m * 64 + kb2);
      const v16bf kf01 = ld_contig(lkb + m * 64 + 32 + kb2);
      const v16bf kf10 = ld_contig(lkb + (16 + m) * 64 + kb2);
      const v16bf kf11 = ld_contig(lkb + (16 + m) * 64 + 32 + kb2);
      v8f s0 = __builtin_amdgcn_wmma_f32_16x16x32_bf16(false, qa0, false, kf00, (short)0, zero8(), false, false);
      s0 = __builtin_amdgcn_wmma_f32_16x16x32_bf16(false, qa1, false, kf01, (short)0, s0, false, false);
      v8f s1 = __builtin_amdgcn_wmma_f32_16x16x32_bf16(false, qa0, false, kf10, (short)0, zero8(), false, false);
      s1 = __builtin_amdgcn_wmma_f32_16x16x32_bf16(false, qa1, false, kf11, (short)0, s1, false, false);

      // band mask (element: query i = iw + r + 8*hi, key j = j0 + ntile*16 + m)
#pragma unroll
      for (int r = 0; r < 8; ++r) {
        const int i = iw + r + hi * 8;
        const int jA = j0 + m, jB = j0 + 16 + m;
        const int dA = (i > jA) ? (i - jA) : (jA - i);
        const int dB = (i > jB) ? (i - jB) : (jB - i);
        if (dA > WIN) s0[r] = -3.0e38f;
        if (dB > WIN) s1[r] = -3.0e38f;
      }

      // online softmax: row stats across 16 lanes (N dimension)
      float pm[8], ps[8], corr[8];
#pragma unroll
      for (int r = 0; r < 8; ++r) pm[r] = fmaxf(s0[r], s1[r]);
#pragma unroll
      for (int d = 1; d < 16; d <<= 1)
#pragma unroll
        for (int r = 0; r < 8; ++r) pm[r] = fmaxf(pm[r], __shfl_xor(pm[r], d, 32));
#pragma unroll
      for (int r = 0; r < 8; ++r) {
        const float nm = fmaxf(rm[r], pm[r]);
        corr[r] = __expf(rm[r] - nm);
        const float p0 = (s0[r] < -1.0e37f) ? 0.0f : __expf(s0[r] - nm);
        const float p1 = (s1[r] < -1.0e37f) ? 0.0f : __expf(s1[r] - nm);
        rm[r] = nm; s0[r] = p0; s1[r] = p1; ps[r] = p0 + p1;
      }
#pragma unroll
      for (int d = 1; d < 16; d <<= 1)
#pragma unroll
        for (int r = 0; r < 8; ++r) ps[r] += __shfl_xor(ps[r], d, 32);
#pragma unroll
      for (int r = 0; r < 8; ++r) {
        rs[r] = rs[r] * corr[r] + ps[r];
        o0[r] *= corr[r]; o1[r] *= corr[r]; o2[r] *= corr[r]; o3[r] *= corr[r];
      }

      // bounce P (C/D layout) through per-wave LDS into A-fragment layout
      unsigned short* pw = &lp[wave][0];
#pragma unroll
      for (int r = 0; r < 8; ++r) {
        const int m2 = r + hi * 8;
        pw[m2 * 32 + m]      = f2bf(s0[r]);
        pw[m2 * 32 + 16 + m] = f2bf(s1[r]);
      }
      const v16bf pa = ld_split(pw + m * 32 + kb2, pw + m * 32 + kb2 + 8);

      const unsigned short* lvb = &lv[buf][0];
      const v16bf vb0 = ld_contig(lvb + m * 32 + kb2);
      const v16bf vb1 = ld_contig(lvb + (16 + m) * 32 + kb2);
      const v16bf vb2 = ld_contig(lvb + (32 + m) * 32 + kb2);
      const v16bf vb3 = ld_contig(lvb + (48 + m) * 32 + kb2);
      o0 = __builtin_amdgcn_wmma_f32_16x16x32_bf16(false, pa, false, vb0, (short)0, o0, false, false);
      o1 = __builtin_amdgcn_wmma_f32_16x16x32_bf16(false, pa, false, vb1, (short)0, o1, false, false);
      o2 = __builtin_amdgcn_wmma_f32_16x16x32_bf16(false, pa, false, vb2, (short)0, o2, false, false);
      o3 = __builtin_amdgcn_wmma_f32_16x16x32_bf16(false, pa, false, vb3, (short)0, o3, false, false);
    }

    wait_asynccnt0();
    __syncthreads();
  }

  const size_t ob = (size_t)h * HEAD_DIM;
#pragma unroll
  for (int r = 0; r < 8; ++r) {
    const int i = iw + r + hi * 8;
    const float inv = 1.0f / rs[r];
    float* orow = out + (size_t)i * D_MODEL + ob + m;
    orow[0]  = o0[r] * inv;
    orow[16] = o1[r] * inv;
    orow[32] = o2[r] * inv;
    orow[48] = o3[r] * inv;
  }
}

// ---------------------------------------------------------------------------
extern "C" void kernel_launch(void* const* d_in, const int* in_sizes, int n_in,
                              void* d_out, int out_size, void* d_ws, size_t ws_size,
                              hipStream_t stream) {
  const float* x  = (const float*)d_in[0];
  const float* Wq = (const float*)d_in[1];
  const float* bq = (const float*)d_in[2];
  const float* Wk = (const float*)d_in[3];
  const float* bk = (const float*)d_in[4];
  const float* Wv = (const float*)d_in[5];
  const float* bv = (const float*)d_in[6];
  float* out = (float*)d_out;

  // workspace layout (bf16 halves): xb | wb | Qb | Kb | Vt  (~38 MB total)
  unsigned short* xb = (unsigned short*)d_ws;
  unsigned short* wb = xb + (size_t)S_LEN * D_MODEL;
  unsigned short* Qb = wb + (size_t)3 * D_MODEL * D_MODEL;
  unsigned short* Kb = Qb + (size_t)N_HEADS * S_LEN * HEAD_DIM;
  unsigned short* Vt = Kb + (size_t)N_HEADS * S_LEN * HEAD_DIM;

  const long long total = (long long)S_LEN * D_MODEL + 3LL * D_MODEL * D_MODEL;
  pack_kernel<<<dim3((unsigned)((total + 255) / 256)), dim3(256), 0, stream>>>(
      x, Wq, Wk, Wv, xb, wb);
  qkv_gemm_kernel<<<dim3(3 * D_MODEL / 64, S_LEN / 128), dim3(256), 0, stream>>>(
      xb, wb, bq, bk, bv, Qb, Kb, Vt);
  attn_kernel<<<dim3(S_LEN / 128, N_HEADS), dim3(256), 0, stream>>>(Qb, Kb, Vt, out);
}
